// GMamba_481036337188
// MI455X (gfx1250) — matrix operand order repaired and checked by reading, not compiled, not run on hardware
//
#include <hip/hip_runtime.h>
#include <hip/hip_bf16.h>

// ---------------------------------------------------------------------------
// GMamba on MI455X (gfx1250): bf16 WMMA GEMMs + wave32 LN + selective scan
// ---------------------------------------------------------------------------

#define DD   128        // hidden dim
#define TT   32         // time steps
#define HWX  256        // H*W
#define NTOK 65536      // B*T*H*W tokens
#define LNW  136        // padded LDS row stride in bf16 elements (272B, 16B aligned)

typedef __attribute__((ext_vector_type(16))) __bf16 v16bf;
typedef __attribute__((ext_vector_type(8)))  float  v8f;

union AFrag {
    v16bf v;
    uint4 q[2];
};

static __device__ __forceinline__ unsigned short f2bf(float f) {
    unsigned u = __builtin_bit_cast(unsigned, f);
    unsigned rnd = 0x7FFFu + ((u >> 16) & 1u);   // round-to-nearest-even
    return (unsigned short)((u + rnd) >> 16);
}
static __device__ __forceinline__ float bf2f(unsigned short h) {
    unsigned u = ((unsigned)h) << 16;
    return __builtin_bit_cast(float, u);
}

static __device__ __forceinline__ v8f wmma_bf16(const AFrag& a, const AFrag& b,
                                                v8f c) {
    return __builtin_amdgcn_wmma_f32_16x16x32_bf16(
        false, a.v, false, b.v, (short)0, c, false, false);
}

// ---------------------------------------------------------------------------
// Kernel 0: repack W (f32 row-major [k][n], 128x128) into WMMA B-fragment
// order as bf16 pairs. Layout: dword index = ((nt*4 + kb)*32 + lane)*8 + v
// where lane = 16*c + n, and dword v packs W[k][n],W[k+1][n] with
// k = kb*32 + 16*c + 2*v   (contiguous k order, same as the A fragments).
// ---------------------------------------------------------------------------
__global__ __launch_bounds__(256) void k_swz(const float* __restrict__ WB,
                                             const float* __restrict__ WC,
                                             const float* __restrict__ Wo,
                                             unsigned* __restrict__ out) {
    int g = blockIdx.x * 256 + threadIdx.x;          // 0 .. 3*8192-1
    if (g >= 3 * 8192) return;
    int matrix = g / 8192;
    int r      = g % 8192;
    int v      = r & 7;
    int lane   = (r >> 3) & 31;
    int kb     = (r >> 8) & 3;
    int nt     = r >> 10;
    const float* W = (matrix == 0) ? WB : (matrix == 1) ? WC : Wo;
    int n = nt * 16 + (lane & 15);
    int c = lane >> 4;
    int k = kb * 32 + 16 * c + 2 * v;
    unsigned lo = f2bf(W[k * DD + n]);
    unsigned hi = f2bf(W[(k + 1) * DD + n]);
    out[g] = lo | (hi << 16);
}

static __device__ __forceinline__ void wave_sum2(float& s, float& ss) {
#pragma unroll
    for (int o = 16; o; o >>= 1) {
        s  += __shfl_xor(s,  o, 32);
        ss += __shfl_xor(ss, o, 32);
    }
}

// ---------------------------------------------------------------------------
// Kernel 1: per 128-token tile: d = x_t - x_{t-1}; LN1; LN2; bf16 -> LDS;
// then GEMMs Bt = xn@WB + bB (f32 out) and Ct = xn@WC + bC (bf16 out)
// using v_wmma_f32_16x16x32_bf16. 8 waves, wave w owns m-tile w.
// Two n-tiles in flight per iteration to hide WMMA RAW hazards.
// ---------------------------------------------------------------------------
__global__ __launch_bounds__(256) void k_ln_proj(
    const float* __restrict__ x,
    const float* __restrict__ g1, const float* __restrict__ b1,
    const float* __restrict__ g2, const float* __restrict__ b2,
    const float* __restrict__ bB, const float* __restrict__ bC,
    const uint4* __restrict__ wBf, const uint4* __restrict__ wCf,
    float* __restrict__ Bt, unsigned short* __restrict__ Ct) {
    __shared__ __align__(16) unsigned short s_xn[128 * LNW];

    const int tid  = threadIdx.x;
    const int wv   = tid >> 5;
    const int lane = tid & 31;
    const int base = blockIdx.x * 128;

    // Warm the 64KB of WB/WC fragments toward this WGP while LN runs.
    {
        const char* wpre = (const char*)wBf;
#pragma unroll
        for (int j = 0; j < 2; ++j)
            __builtin_prefetch(wpre + (size_t)(j * 256 + tid) * 128, 0, 3);
    }

    const int ch0 = lane * 4;
    const float4 vg1 = *(const float4*)(g1 + ch0);
    const float4 vb1 = *(const float4*)(b1 + ch0);
    const float4 vg2 = *(const float4*)(g2 + ch0);
    const float4 vb2 = *(const float4*)(b2 + ch0);

    // ---- Stage A: diff + double LayerNorm, 16 tokens per wave -------------
#pragma unroll 1
    for (int i = 0; i < 16; ++i) {
        int row = wv * 16 + i;
        int tok = base + row;
        int t   = (tok >> 8) & (TT - 1);
        float4 xv = *(const float4*)(x + (size_t)tok * DD + ch0);
        float4 xp = make_float4(0.f, 0.f, 0.f, 0.f);
        if (t != 0) xp = *(const float4*)(x + (size_t)(tok - HWX) * DD + ch0);
        float d0 = xv.x - xp.x, d1 = xv.y - xp.y, d2 = xv.z - xp.z, d3 = xv.w - xp.w;

        float s  = d0 + d1 + d2 + d3;
        float ss = d0 * d0 + d1 * d1 + d2 * d2 + d3 * d3;
        wave_sum2(s, ss);
        float mu = s * (1.f / 128.f);
        float rs = rsqrtf(ss * (1.f / 128.f) - mu * mu + 1e-5f);
        float n0 = (d0 - mu) * rs * vg1.x + vb1.x;
        float n1 = (d1 - mu) * rs * vg1.y + vb1.y;
        float n2 = (d2 - mu) * rs * vg1.z + vb1.z;
        float n3 = (d3 - mu) * rs * vg1.w + vb1.w;

        s  = n0 + n1 + n2 + n3;
        ss = n0 * n0 + n1 * n1 + n2 * n2 + n3 * n3;
        wave_sum2(s, ss);
        mu = s * (1.f / 128.f);
        rs = rsqrtf(ss * (1.f / 128.f) - mu * mu + 1e-5f);
        float e0 = (n0 - mu) * rs * vg2.x + vb2.x;
        float e1 = (n1 - mu) * rs * vg2.y + vb2.y;
        float e2 = (n2 - mu) * rs * vg2.z + vb2.z;
        float e3 = (n3 - mu) * rs * vg2.w + vb2.w;

        unsigned lo = (unsigned)f2bf(e0) | ((unsigned)f2bf(e1) << 16);
        unsigned hi = (unsigned)f2bf(e2) | ((unsigned)f2bf(e3) << 16);
        *(uint2*)&s_xn[row * LNW + ch0] = make_uint2(lo, hi);
    }
    __syncthreads();

    // ---- Stage B: dual GEMM with WMMA -------------------------------------
    const int c  = lane >> 4;
    const int nn = lane & 15;
    // A fragments (contiguous k order, matching the B repack): lane holds one contiguous 16-bf16
    // run: K = kb*32 + c*16 + [0..15]
    AFrag a[4];
    const int arow = wv * 16 + nn;
#pragma unroll
    for (int kb = 0; kb < 4; ++kb) {
        const unsigned short* p = &s_xn[arow * LNW + kb * 32 + c * 16];
        a[kb].q[0] = *(const uint4*)p;
        a[kb].q[1] = *(const uint4*)(p + 8);
    }

#pragma unroll 1
    for (int mat = 0; mat < 2; ++mat) {
        const uint4* wf = mat ? wCf : wBf;
        const float* bias = mat ? bC : bB;
#pragma unroll 1
        for (int nt = 0; nt < 8; nt += 2) {
            v8f acc0 = {0.f, 0.f, 0.f, 0.f, 0.f, 0.f, 0.f, 0.f};
            v8f acc1 = {0.f, 0.f, 0.f, 0.f, 0.f, 0.f, 0.f, 0.f};
#pragma unroll
            for (int kb = 0; kb < 4; ++kb) {
                AFrag b0, b1;
                const uint4* s0 = wf + (size_t)(((nt * 4 + kb) * 32 + lane) * 2);
                const uint4* s1 = wf + (size_t)((((nt + 1) * 4 + kb) * 32 + lane) * 2);
                b0.q[0] = s0[0]; b0.q[1] = s0[1];
                b1.q[0] = s1[0]; b1.q[1] = s1[1];
                acc0 = wmma_bf16(a[kb], b0, acc0);
                acc1 = wmma_bf16(a[kb], b1, acc1);
            }
            float bv0 = bias[nt * 16 + nn];
            float bv1 = bias[nt * 16 + 16 + nn];
            if (mat == 0) {
#pragma unroll
                for (int i = 0; i < 8; ++i) {
                    int m   = i + 8 * c;
                    int tok = base + wv * 16 + m;
                    Bt[(size_t)tok * DD + nt * 16 + nn]      = acc0[i] + bv0;
                    Bt[(size_t)tok * DD + nt * 16 + 16 + nn] = acc1[i] + bv1;
                }
            } else {
#pragma unroll
                for (int i = 0; i < 8; ++i) {
                    int m   = i + 8 * c;
                    int tok = base + wv * 16 + m;
                    Ct[(size_t)tok * DD + nt * 16 + nn]      = f2bf(acc0[i] + bv0);
                    Ct[(size_t)tok * DD + nt * 16 + 16 + nn] = f2bf(acc1[i] + bv1);
                }
            }
        }
    }
}

// ---------------------------------------------------------------------------
// Kernel 2: selective scan  h_t = exp(A_d)*h_{t-1} + Bt ;  y = Ct * h
// One thread per (b, hw, d); coalesced over d.
// ---------------------------------------------------------------------------
__global__ __launch_bounds__(256) void k_scan(const float* __restrict__ Bt,
                                              const unsigned short* __restrict__ Ct,
                                              const float* __restrict__ A,
                                              unsigned short* __restrict__ y) {
    int g  = blockIdx.x * 256 + threadIdx.x;   // 0 .. 262143
    int d  = g & (DD - 1);
    int hw = (g >> 7) & (HWX - 1);
    int b  = g >> 15;
    float decay = expf(A[d]);
    size_t idx = (((size_t)b * TT) * HWX + hw) * DD + d;
    float h = 0.f;
#pragma unroll 1
    for (int t = 0; t < TT; ++t) {
        h = decay * h + Bt[idx];
        y[idx] = f2bf(bf2f(Ct[idx]) * h);
        idx += (size_t)HWX * DD;
    }
}

// ---------------------------------------------------------------------------
// Kernel 3: out = d_seq + y @ Wo + bo   (y staged in LDS, WMMA GEMM,
// d_seq recomputed from x in the epilogue to avoid a 32MB buffer).
// ---------------------------------------------------------------------------
__global__ __launch_bounds__(256) void k_out(const unsigned short* __restrict__ y,
                                             const uint4* __restrict__ wOf,
                                             const float* __restrict__ bo,
                                             const float* __restrict__ x,
                                             float* __restrict__ out) {
    __shared__ __align__(16) unsigned short s_y[128 * LNW];
    const int tid  = threadIdx.x;
    const int wv   = tid >> 5;
    const int lane = tid & 31;
    const int base = blockIdx.x * 128;

    // Warm the 32KB of Wo fragments toward this WGP while staging y.
    __builtin_prefetch((const char*)wOf + (size_t)tid * 128, 0, 3);

    // stage 128x128 bf16 y tile into LDS, 16B chunks, coalesced
#pragma unroll
    for (int j = 0; j < 8; ++j) {
        int e   = (j * 256 + tid) * 8;        // bf16 element index in tile
        int row = e >> 7;
        int col = e & (DD - 1);
        *(uint4*)&s_y[row * LNW + col] =
            *(const uint4*)(y + (size_t)(base + row) * DD + col);
    }
    __syncthreads();

    const int c  = lane >> 4;
    const int nn = lane & 15;
    AFrag a[4];
    const int arow = wv * 16 + nn;
#pragma unroll
    for (int kb = 0; kb < 4; ++kb) {
        const unsigned short* p = &s_y[arow * LNW + kb * 32 + c * 16];
        a[kb].q[0] = *(const uint4*)p;
        a[kb].q[1] = *(const uint4*)(p + 8);
    }

#pragma unroll 1
    for (int nt = 0; nt < 8; nt += 2) {
        v8f acc0 = {0.f, 0.f, 0.f, 0.f, 0.f, 0.f, 0.f, 0.f};
        v8f acc1 = {0.f, 0.f, 0.f, 0.f, 0.f, 0.f, 0.f, 0.f};
#pragma unroll
        for (int kb = 0; kb < 4; ++kb) {
            AFrag b0, b1;
            const uint4* s0 = wOf + (size_t)(((nt * 4 + kb) * 32 + lane) * 2);
            const uint4* s1 = wOf + (size_t)((((nt + 1) * 4 + kb) * 32 + lane) * 2);
            b0.q[0] = s0[0]; b0.q[1] = s0[1];
            b1.q[0] = s1[0]; b1.q[1] = s1[1];
            acc0 = wmma_bf16(a[kb], b0, acc0);
            acc1 = wmma_bf16(a[kb], b1, acc1);
        }
        float bv0  = bo[nt * 16 + nn];
        float bv1  = bo[nt * 16 + 16 + nn];
        int   col0 = nt * 16 + nn;
        int   col1 = nt * 16 + 16 + nn;
#pragma unroll
        for (int i = 0; i < 8; ++i) {
            int m   = i + 8 * c;
            int tok = base + wv * 16 + m;
            int t   = (tok >> 8) & (TT - 1);
            float xv0 = x[(size_t)tok * DD + col0];
            float xv1 = x[(size_t)tok * DD + col1];
            float xp0 = (t != 0) ? x[(size_t)(tok - HWX) * DD + col0] : 0.f;
            float xp1 = (t != 0) ? x[(size_t)(tok - HWX) * DD + col1] : 0.f;
            out[(size_t)tok * DD + col0] = acc0[i] + bv0 + (xv0 - xp0);
            out[(size_t)tok * DD + col1] = acc1[i] + bv1 + (xv1 - xp1);
        }
    }
}

// ---------------------------------------------------------------------------
// Host-side launcher
// ---------------------------------------------------------------------------
extern "C" void kernel_launch(void* const* d_in, const int* in_sizes, int n_in,
                              void* d_out, int out_size, void* d_ws, size_t ws_size,
                              hipStream_t stream) {
    (void)in_sizes; (void)n_in; (void)out_size; (void)ws_size;
    const float* x  = (const float*)d_in[0];
    const float* g1 = (const float*)d_in[1];
    const float* b1 = (const float*)d_in[2];
    const float* g2 = (const float*)d_in[3];
    const float* b2 = (const float*)d_in[4];
    const float* WB = (const float*)d_in[5];
    const float* bB = (const float*)d_in[6];
    const float* WC = (const float*)d_in[7];
    const float* bC = (const float*)d_in[8];
    const float* Wo = (const float*)d_in[9];
    const float* bo = (const float*)d_in[10];
    const float* A  = (const float*)d_in[11];
    float* outp = (float*)d_out;

    // workspace layout
    char* ws = (char*)d_ws;
    unsigned*       wfrag = (unsigned*)ws;                         // 96 KB (3 * 8192 dwords)
    float*          Bt    = (float*)(ws + (128 << 10));            // 32 MB
    unsigned short* Ct    = (unsigned short*)((char*)Bt + (size_t)NTOK * DD * 4);   // 16 MB
    unsigned short* yb    = (unsigned short*)((char*)Ct + (size_t)NTOK * DD * 2);   // 16 MB

    const uint4* wBf = (const uint4*)wfrag;
    const uint4* wCf = wBf + 2048;
    const uint4* wOf = wBf + 4096;

    k_swz<<<96, 256, 0, stream>>>(WB, WC, Wo, wfrag);
    k_ln_proj<<<NTOK / 128, 256, 0, stream>>>(x, g1, b1, g2, b2, bB, bC,
                                              wBf, wCf, Bt, Ct);
    k_scan<<<(8 * HWX * DD) / 256, 256, 0, stream>>>(Bt, Ct, A, yb);
    k_out<<<NTOK / 128, 256, 0, stream>>>(yb, wOf, bo, x, outp);
}
